// SpatioTemporalTransformerLayer_68255620268644
// MI455X (gfx1250) — compile-verified, matrix operand order checked by reading, not run on hardware
//
#include <hip/hip_runtime.h>

// ---------------------------------------------------------------------------
// SpatioTemporalTransformerLayer for MI455X (gfx1250, wave32, WMMA).
// B=4, S=1024, D=1024, H=16, HD=64, DFF=4096, BIAS_HID=8.
// All GEMMs use v_wmma_f32_16x16x32_bf16 (bf16 A/B, f32 accumulate).
// GEMM K-loop is software-pipelined with ping-pong fragment buffers
// (2x unrolled) so global loads overlap WMMAs with no register copies.
// ---------------------------------------------------------------------------

typedef __attribute__((ext_vector_type(16))) __bf16 v16bf;
typedef __attribute__((ext_vector_type(8)))  __bf16 v8bf;
typedef __attribute__((ext_vector_type(8)))  float  v8f;

#define BDIM 4
#define SDIM 1024
#define DMODEL 1024
#define NHEAD 16
#define HEADD 64
#define DFF 4096

__device__ __forceinline__ __bf16 f2bf(float x) {
    unsigned u = __builtin_bit_cast(unsigned, x);
    unsigned r = (u + 0x7FFFu + ((u >> 16) & 1u)) >> 16;
    return __builtin_bit_cast(__bf16, (unsigned short)r);
}

__device__ __forceinline__ v8f wmma_bf16(v16bf a, v16bf b, v8f c) {
    // (neg_a, A, neg_b, B, c_mod, C, reuse_a, reuse_b)
    return __builtin_amdgcn_wmma_f32_16x16x32_bf16(false, a, false, b, (short)0, c,
                                                   false, false);
}

// Load a 16x32 bf16 A/B fragment: lane L, elem e -> (row=L&15, K=16*(e>>3)+8*(L>>4)+(e&7))
// 'row_ptr' points at element K=0 of this lane's row. Two contiguous 16B loads.
__device__ __forceinline__ v16bf load_frag(const __bf16* row_ptr, int hl) {
    v8bf lo = *(const v8bf*)(row_ptr + hl * 8);
    v8bf hi = *(const v8bf*)(row_ptr + 16 + hl * 8);
    v16bf f;
#pragma unroll
    for (int i = 0; i < 8; ++i) { f[i] = lo[i]; f[8 + i] = hi[i]; }
    return f;
}

// ---------------------------------------------------------------------------
// Elementwise converters
// ---------------------------------------------------------------------------
__global__ void cvt_f32_bf16_kernel(const float* __restrict__ in,
                                    __bf16* __restrict__ out, int n) {
    int i = blockIdx.x * 256 + threadIdx.x;
    if (i < n) out[i] = f2bf(in[i]);
}

// W[K][N] f32 (row-major) -> Wt[N][K] bf16
__global__ void cvt_transpose_kernel(const float* __restrict__ W,
                                     __bf16* __restrict__ Wt, int K, int N) {
    int i = blockIdx.x * 256 + threadIdx.x;
    if (i < K * N) {
        int k = i / N, n = i % N;
        Wt[(size_t)n * K + k] = f2bf(W[i]);
    }
}

// ---------------------------------------------------------------------------
// Generic bf16 WMMA GEMM: C[M,N] = A[M,K] * Wt[N,K]^T + bias, with mode epilogue
//   mode 0: Q/K layout  -> bf16 out[((b*16+h)*1024+s)*64+d]
//   mode 2: V transposed-> bf16 out[((b*16+h)*64+d)*1024+s]
//   mode 3: ReLU        -> bf16 out[row*N+col]   (FFN1)
//   mode 4: plain f32   -> f32  out[row*N+col]   (FFN2)
// Block: 128 threads = 4 waves; wave w owns rows m0+32w..+31, cols n0..n0+63.
// Per k-chunk: 2 A frags + 4 B frags -> 8 WMMAs; ping-pong double buffering.
// ---------------------------------------------------------------------------
__global__ __launch_bounds__(128) void gemm_bf16_kernel(
    const __bf16* __restrict__ A, const __bf16* __restrict__ Wt,
    const float* __restrict__ bias, void* __restrict__ outp,
    int N, int Kd, int mode) {
    const int lane = threadIdx.x & 31;
    const int wv = threadIdx.x >> 5;
    const int hl = lane >> 4;
    const int l16 = lane & 15;
    const int n0 = blockIdx.x * 64;
    const int m0 = blockIdx.y * 128 + wv * 32;

    v8f c[2][4] = {};
    const __bf16* arow0 = A + (size_t)(m0 + l16) * Kd;
    const __bf16* arow1 = A + (size_t)(m0 + 16 + l16) * Kd;
    const __bf16* brow[4];
#pragma unroll
    for (int g = 0; g < 4; ++g) brow[g] = Wt + (size_t)(n0 + g * 16 + l16) * Kd;

    // ping-pong fragment buffers
    v16bf a0[2], a1[2], bb[2][4];

#define LOAD_CHUNK(buf, kk)                                        \
    do {                                                           \
        a0[buf] = load_frag(arow0 + (kk), hl);                     \
        a1[buf] = load_frag(arow1 + (kk), hl);                     \
        _Pragma("unroll") for (int g = 0; g < 4; ++g)              \
            bb[buf][g] = load_frag(brow[g] + (kk), hl);            \
    } while (0)

#define COMPUTE(buf)                                               \
    do {                                                           \
        _Pragma("unroll") for (int g = 0; g < 4; ++g) {            \
            c[0][g] = wmma_bf16(a0[buf], bb[buf][g], c[0][g]);     \
            c[1][g] = wmma_bf16(a1[buf], bb[buf][g], c[1][g]);     \
        }                                                          \
    } while (0)

    // Kd is a multiple of 64 (1024 or 4096) -> even number of 32-wide chunks.
    LOAD_CHUNK(0, 0);
    LOAD_CHUNK(1, 32);
    for (int kc = 0; kc < Kd - 64; kc += 64) {
        COMPUTE(0);
        LOAD_CHUNK(0, kc + 64);
        COMPUTE(1);
        LOAD_CHUNK(1, kc + 96);
    }
    COMPUTE(0);
    COMPUTE(1);
#undef LOAD_CHUNK
#undef COMPUTE

#pragma unroll
    for (int sub = 0; sub < 2; ++sub) {
#pragma unroll
        for (int g = 0; g < 4; ++g) {
            const int col = n0 + g * 16 + l16;
            const float bcst = bias[col];
#pragma unroll
            for (int v = 0; v < 8; ++v) {
                const int row = m0 + sub * 16 + v + hl * 8;  // D-frag: M = v + 8*(lane>=16)
                float val = c[sub][g][v] + bcst;
                if (mode == 3) {
                    val = fmaxf(val, 0.f);
                    ((__bf16*)outp)[(size_t)row * N + col] = f2bf(val);
                } else if (mode == 4) {
                    ((float*)outp)[(size_t)row * N + col] = val;
                } else if (mode == 2) {  // V -> [b,h,d,s]
                    int bb2 = row >> 10, s = row & 1023, hh = col >> 6, d = col & 63;
                    ((__bf16*)outp)[(((size_t)(bb2 * 16 + hh)) * 64 + d) * 1024 + s] = f2bf(val);
                } else {                 // Q/K -> [b,h,s,d]
                    int bb2 = row >> 10, s = row & 1023, hh = col >> 6, d = col & 63;
                    ((__bf16*)outp)[(((size_t)(bb2 * 16 + hh)) * 1024 + s) * 64 + d] = f2bf(val);
                }
            }
        }
    }
}

// ---------------------------------------------------------------------------
// Flash attention with fused pair-bias MLP. One wave per (b, h, 16-row q-tile).
// ---------------------------------------------------------------------------
__global__ __launch_bounds__(32) void attn_kernel(
    const __bf16* __restrict__ Q, const __bf16* __restrict__ Km,
    const __bf16* __restrict__ Vt, const float* __restrict__ td,
    const float* __restrict__ dd, const float* __restrict__ tw1,
    const float* __restrict__ tb1, const float* __restrict__ tw2,
    const float* __restrict__ tb2, const float* __restrict__ dw1,
    const float* __restrict__ db1, const float* __restrict__ dw2,
    const float* __restrict__ db2, float* __restrict__ out) {
    const int lane = threadIdx.x & 31;
    const int hl = lane >> 4;
    const int l16 = lane & 15;
    const int qt = blockIdx.x & 63;
    const int h = (blockIdx.x >> 6) & 15;
    const int b = blockIdx.x >> 10;

    __shared__ float sc[16][36];                                // f32 score tile 16x32 (+pad)
    __shared__ __attribute__((aligned(16))) __bf16 pb[16][40];  // bf16 P tile
    __shared__ float corr[16];
    __shared__ float lsum[16];

    // Per-head pair-bias MLP weights (uniform -> scalar loads)
    float w1t[8], b1t[8], w2t[8], w1d[8], b1d[8], w2d[8];
#pragma unroll
    for (int j = 0; j < 8; ++j) {
        w1t[j] = tw1[j]; b1t[j] = tb1[j]; w2t[j] = tw2[j * 16 + h];
        w1d[j] = dw1[j]; b1d[j] = db1[j]; w2d[j] = dw2[j * 16 + h];
    }
    const float bconst = tb2[h] + db2[h];

    const size_t bh = (size_t)(b * 16 + h);

    // Q A-fragments for the two K-dim halves (hd 0..31, 32..63)
    v16bf qa[2];
    {
        const __bf16* r = Q + (bh * 1024 + qt * 16 + l16) * 64;
        qa[0] = load_frag(r, hl);
        qa[1] = load_frag(r + 32, hl);
    }

    v8f o[4] = {};
    float mrow = -1e30f, lrow = 0.f;

    for (int kb = 0; kb < SDIM; kb += 32) {
        // ---- scores for 32 keys: 2 N-groups x 2 K-frags = 4 WMMAs ----
#pragma unroll
        for (int g = 0; g < 2; ++g) {
            const __bf16* kr = Km + (bh * 1024 + kb + g * 16 + l16) * 64;
            v16bf kb0 = load_frag(kr, hl);
            v16bf kb1 = load_frag(kr + 32, hl);
            v8f s = {};
            s = wmma_bf16(qa[0], kb0, s);
            s = wmma_bf16(qa[1], kb1, s);
            const int key = kb + g * 16 + l16;
#pragma unroll
            for (int v = 0; v < 8; ++v) {
                const int qrow = v + hl * 8;
                const int qg = qt * 16 + qrow;
                const size_t midx = ((size_t)b * SDIM + qg) * SDIM + key;
                const float m1 = td[midx];
                const float m2 = dd[midx];
                float bias = bconst;
#pragma unroll
                for (int j = 0; j < 8; ++j) {
                    bias += fmaxf(m1 * w1t[j] + b1t[j], 0.f) * w2t[j];
                    bias += fmaxf(m2 * w1d[j] + b1d[j], 0.f) * w2d[j];
                }
                sc[qrow][g * 16 + l16] = s[v] * (1.0f / 32.0f) + bias;
            }
        }
        __syncthreads();

        // ---- online softmax: lane handles (row=l16, cols hl*16..+15) ----
        float pm = -1e30f;
        float sv[16];
#pragma unroll
        for (int j = 0; j < 16; ++j) {
            sv[j] = sc[l16][hl * 16 + j];
            pm = fmaxf(pm, sv[j]);
        }
        pm = fmaxf(pm, __shfl_xor(pm, 16, 32));
        const float nm = fmaxf(mrow, pm);
        const float cf = __expf(mrow - nm);
        float ps = 0.f;
#pragma unroll
        for (int j = 0; j < 16; ++j) {
            float p = __expf(sv[j] - nm);
            ps += p;
            pb[l16][hl * 16 + j] = f2bf(p);
        }
        ps += __shfl_xor(ps, 16, 32);
        lrow = lrow * cf + ps;
        mrow = nm;
        if (lane < 16) corr[l16] = cf;
        __syncthreads();

        // ---- rescale O, then O += P * V (4 WMMAs) ----
        float cfa[8];
#pragma unroll
        for (int v = 0; v < 8; ++v) cfa[v] = corr[v + hl * 8];
#pragma unroll
        for (int g = 0; g < 4; ++g)
#pragma unroll
            for (int v = 0; v < 8; ++v) o[g][v] *= cfa[v];

        v16bf pa = load_frag(&pb[l16][0], hl);
#pragma unroll
        for (int g = 0; g < 4; ++g) {
            const __bf16* vr = Vt + (bh * 64 + g * 16 + l16) * 1024 + kb;
            v16bf vb = load_frag(vr, hl);
            o[g] = wmma_bf16(pa, vb, o[g]);
        }
        __syncthreads();  // sc/pb reused next iteration
    }

    if (lane < 16) lsum[l16] = 1.f / lrow;
    __syncthreads();
    float ri[8];
#pragma unroll
    for (int v = 0; v < 8; ++v) ri[v] = lsum[v + hl * 8];
#pragma unroll
    for (int g = 0; g < 4; ++g)
#pragma unroll
        for (int v = 0; v < 8; ++v) {
            const int s = qt * 16 + v + hl * 8;
            const int d = h * 64 + g * 16 + l16;
            out[((size_t)b * SDIM + s) * DMODEL + d] = o[g][v] * ri[v];
        }
}

// ---------------------------------------------------------------------------
// Residual + LayerNorm over D=1024. One 256-thread block per row.
// ---------------------------------------------------------------------------
__global__ __launch_bounds__(256) void add_ln_kernel(
    const float* __restrict__ a, const float* __restrict__ r,
    const float* __restrict__ g, const float* __restrict__ beta,
    float* __restrict__ outf, __bf16* __restrict__ outb) {
    const int row = blockIdx.x;
    const int t = threadIdx.x;
    __shared__ float red[256];
    float v[4];
    float s = 0.f;
#pragma unroll
    for (int i = 0; i < 4; ++i) {
        const int c = t + i * 256;
        v[i] = a[(size_t)row * DMODEL + c] + r[(size_t)row * DMODEL + c];
        s += v[i];
    }
    red[t] = s;
    __syncthreads();
    for (int st = 128; st > 0; st >>= 1) {
        if (t < st) red[t] += red[t + st];
        __syncthreads();
    }
    const float mean = red[0] * (1.f / 1024.f);
    __syncthreads();
    float ss = 0.f;
#pragma unroll
    for (int i = 0; i < 4; ++i) {
        const float d = v[i] - mean;
        ss += d * d;
    }
    red[t] = ss;
    __syncthreads();
    for (int st = 128; st > 0; st >>= 1) {
        if (t < st) red[t] += red[t + st];
        __syncthreads();
    }
    const float rstd = rsqrtf(red[0] * (1.f / 1024.f) + 1e-5f);
#pragma unroll
    for (int i = 0; i < 4; ++i) {
        const int c = t + i * 256;
        const float y = (v[i] - mean) * rstd * g[c] + beta[c];
        outf[(size_t)row * DMODEL + c] = y;
        if (outb) outb[(size_t)row * DMODEL + c] = f2bf(y);
    }
}

// ---------------------------------------------------------------------------
// Host launch
// ---------------------------------------------------------------------------
extern "C" void kernel_launch(void* const* d_in, const int* in_sizes, int n_in,
                              void* d_out, int out_size, void* d_ws, size_t ws_size,
                              hipStream_t stream) {
    (void)in_sizes; (void)n_in; (void)out_size; (void)ws_size;
    const float* src = (const float*)d_in[0];
    const float* td  = (const float*)d_in[1];
    const float* dd  = (const float*)d_in[2];
    const float* Wq  = (const float*)d_in[3];
    const float* bq  = (const float*)d_in[4];
    const float* Wk  = (const float*)d_in[5];
    const float* bk  = (const float*)d_in[6];
    const float* Wv  = (const float*)d_in[7];
    const float* bv  = (const float*)d_in[8];
    const float* tw1 = (const float*)d_in[9];
    const float* tb1 = (const float*)d_in[10];
    const float* tw2 = (const float*)d_in[11];
    const float* tb2 = (const float*)d_in[12];
    const float* dw1 = (const float*)d_in[13];
    const float* db1 = (const float*)d_in[14];
    const float* dw2 = (const float*)d_in[15];
    const float* db2 = (const float*)d_in[16];
    const float* W1  = (const float*)d_in[17];
    const float* b1  = (const float*)d_in[18];
    const float* W2  = (const float*)d_in[19];
    const float* b2  = (const float*)d_in[20];
    const float* g1  = (const float*)d_in[21];
    const float* be1 = (const float*)d_in[22];
    const float* g2  = (const float*)d_in[23];
    const float* be2 = (const float*)d_in[24];

    const int M = BDIM * SDIM;  // 4096 rows
    char* ws = (char*)d_ws;
    size_t off = 0;
    auto alloc = [&](size_t bytes) {
        char* p = ws + off;
        off = (off + bytes + 255) & ~(size_t)255;
        return p;
    };
    __bf16* srcb = (__bf16*)alloc((size_t)M * DMODEL * 2);
    __bf16* Wqt  = (__bf16*)alloc((size_t)DMODEL * DMODEL * 2);
    __bf16* Wkt  = (__bf16*)alloc((size_t)DMODEL * DMODEL * 2);
    __bf16* Wvt  = (__bf16*)alloc((size_t)DMODEL * DMODEL * 2);
    __bf16* W1t  = (__bf16*)alloc((size_t)DMODEL * DFF * 2);
    __bf16* W2t  = (__bf16*)alloc((size_t)DFF * DMODEL * 2);
    __bf16* Qb   = (__bf16*)alloc((size_t)M * DMODEL * 2);
    __bf16* Kb   = (__bf16*)alloc((size_t)M * DMODEL * 2);
    __bf16* Vtb  = (__bf16*)alloc((size_t)M * DMODEL * 2);
    float*  attn = (float*)alloc((size_t)M * DMODEL * 4);
    float*  xf   = (float*)alloc((size_t)M * DMODEL * 4);
    __bf16* xb   = (__bf16*)alloc((size_t)M * DMODEL * 2);
    __bf16* h1   = (__bf16*)alloc((size_t)M * DFF * 2);
    float*  ffn  = (float*)alloc((size_t)M * DMODEL * 4);

    const int nSrc = M * DMODEL;
    cvt_f32_bf16_kernel<<<(nSrc + 255) / 256, 256, 0, stream>>>(src, srcb, nSrc);
    cvt_transpose_kernel<<<(DMODEL * DMODEL + 255) / 256, 256, 0, stream>>>(Wq, Wqt, DMODEL, DMODEL);
    cvt_transpose_kernel<<<(DMODEL * DMODEL + 255) / 256, 256, 0, stream>>>(Wk, Wkt, DMODEL, DMODEL);
    cvt_transpose_kernel<<<(DMODEL * DMODEL + 255) / 256, 256, 0, stream>>>(Wv, Wvt, DMODEL, DMODEL);
    cvt_transpose_kernel<<<(DMODEL * DFF + 255) / 256, 256, 0, stream>>>(W1, W1t, DMODEL, DFF);
    cvt_transpose_kernel<<<(DFF * DMODEL + 255) / 256, 256, 0, stream>>>(W2, W2t, DFF, DMODEL);

    dim3 gqkv(DMODEL / 64, M / 128);
    gemm_bf16_kernel<<<gqkv, 128, 0, stream>>>(srcb, Wqt, bq, Qb, DMODEL, DMODEL, 0);
    gemm_bf16_kernel<<<gqkv, 128, 0, stream>>>(srcb, Wkt, bk, Kb, DMODEL, DMODEL, 0);
    gemm_bf16_kernel<<<gqkv, 128, 0, stream>>>(srcb, Wvt, bv, Vtb, DMODEL, DMODEL, 2);

    attn_kernel<<<BDIM * NHEAD * (SDIM / 16), 32, 0, stream>>>(
        Qb, Kb, Vtb, td, dd, tw1, tb1, tw2, tb2, dw1, db1, dw2, db2, attn);

    add_ln_kernel<<<M, 256, 0, stream>>>(src, attn, g1, be1, xf, xb);

    gemm_bf16_kernel<<<dim3(DFF / 64, M / 128), 128, 0, stream>>>(xb, W1t, b1, h1, DFF, DMODEL, 3);
    gemm_bf16_kernel<<<dim3(DMODEL / 64, M / 128), 128, 0, stream>>>(h1, W2t, b2, ffn, DMODEL, DFF, 4);

    add_ln_kernel<<<M, 256, 0, stream>>>(xf, ffn, g2, be2, (float*)d_out, nullptr);
}
